// SupervisedContrastiveLoss_25572235281097
// MI455X (gfx1250) — matrix operand order repaired and checked by reading, not compiled, not run on hardware
//
#include <hip/hip_runtime.h>
#include <type_traits>

typedef __attribute__((ext_vector_type(16))) _Float16 v16h;
typedef __attribute__((ext_vector_type(8)))  _Float16 v8h;
typedef __attribute__((ext_vector_type(8)))  float    v8f;

#define DDIM 128
#define TEMP_INV 10.0f
#define EXPK 14.4269504088896f   /* 10 * log2(e): exp(10a-10) = exp2(fma(a,EXPK,-EXPK)) */
#define PTHRESH 0.25f            /* sqrt(|dp|)+1e-8 < 0.5  <=>  |dp| < (0.5-1e-8)^2 ~= 0.25 */
#define LN2 0.693147180559945f

// Raw hardware transcendentals: our exp2 argument range is [-29, 0], far from
// the denormal region, so libm's range-fixup is dead weight.
#if __has_builtin(__builtin_amdgcn_exp2f)
#define EXP2F(x) __builtin_amdgcn_exp2f(x)
#else
#define EXP2F(x) exp2f(x)
#endif
#if __has_builtin(__builtin_amdgcn_logf)
#define LOG2F_RAW(x) __builtin_amdgcn_logf(x)
#else
#define LOG2F_RAW(x) log2f(x)
#endif

// ---------------------------------------------------------------------------
// Kernel 1: L2-normalize each row of the f32 embeddings, store as f16.
// ---------------------------------------------------------------------------
__global__ void __launch_bounds__(128)
scl_norm_f16_kernel(const float* __restrict__ emb, _Float16* __restrict__ eh) {
  const int row = blockIdx.x;
  const int tid = threadIdx.x;
  const float v = emb[(size_t)row * DDIM + tid];
  float sq = v * v;
#pragma unroll
  for (int off = 16; off > 0; off >>= 1) sq += __shfl_xor(sq, off, 32);
  __shared__ float smem[4];
  if ((tid & 31) == 0) smem[tid >> 5] = sq;
  __syncthreads();
  const float total = smem[0] + smem[1] + smem[2] + smem[3];
  const float inv = 1.0f / sqrtf(total);
  eh[(size_t)row * DDIM + tid] = (_Float16)(v * inv);
}

// ---------------------------------------------------------------------------
// Kernel 2: streaming Gram-matrix + fused masked softmax statistics.
// One wave handles 32 rows (2 x 16-row tiles, A held in VGPRs), sweeps all N
// columns in 16-wide tiles: 4x v_wmma_f32_16x16x32_f16 per tile (K=128), then
// a branchless per-lane update of z = sum exp(10a-10), s = sum mask*a,
// c = sum mask. Fixed softmax shift 10 is safe since cos<=1 -> logit<=10.
// The two diagonal-containing tiles (ct = dct0, dct0+1) are PEELED out of the
// sweep so the hot loop carries no diagonal logic or scalar checks at all.
// ---------------------------------------------------------------------------
__global__ void __launch_bounds__(32)
scl_sim_kernel(const _Float16* __restrict__ eh,
               const float* __restrict__ props,
               float* __restrict__ per_row,
               float* __restrict__ validf,
               int n) {
  const int lane = threadIdx.x;
  const int half = lane >> 4;   // which 16-lane half of the wave32
  const int l16  = lane & 15;
  const int rowBase = blockIdx.x * 32;

  // ---- A operands: 2 row-tiles x 4 K-steps, resident for the whole sweep.
  // A k order used here (contiguous, same as B): lanes 0-15 hold M=lane, K=0..15;
  // lanes 16-31 hold M=lane-16, K=16..31. Two 16B chunks per operand.
  v16h A[2][4];
#pragma unroll
  for (int rt = 0; rt < 2; ++rt) {
    const _Float16* rp = eh + (size_t)(rowBase + rt * 16 + l16) * DDIM;
#pragma unroll
    for (int ks = 0; ks < 4; ++ks) {
      const v8h c0 = *(const v8h*)(rp + ks * 32 + (half ? 16 : 0));
      const v8h c1 = *(const v8h*)(rp + ks * 32 + (half ? 24 : 8));
      v16h a;
#pragma unroll
      for (int i = 0; i < 8; ++i) { a[i] = c0[i]; a[i + 8] = c1[i]; }
      A[rt][ks] = a;
    }
  }

  // Row properties for the 8 accumulator slots this lane owns (C-layout rows).
  float prw[2][8];
#pragma unroll
  for (int rt = 0; rt < 2; ++rt) {
    const int base = rowBase + rt * 16 + half * 8;
#pragma unroll
    for (int r = 0; r < 8; ++r) prw[rt][r] = props[base + r];
  }

  // Diagonal ownership: within the diagonal tile, lane l16 == half*8 + r owns
  // the element of accumulator slot r. Each lane owns at most one slot.
  bool laneDiag[8];
#pragma unroll
  for (int r = 0; r < 8; ++r) laneDiag[r] = (l16 == half * 8 + r);
  const int dct0 = rowBase >> 4;       // column-tile holding rt=0's diagonal
  float dval[2] = {0.0f, 0.0f};        // captured sim_ii per row-tile

  float z[2][8], s[2][8], c[2][8];
#pragma unroll
  for (int rt = 0; rt < 2; ++rt)
#pragma unroll
    for (int r = 0; r < 8; ++r) { z[rt][r] = 0.f; s[rt][r] = 0.f; c[rt][r] = 0.f; }

  // One column tile: loads + 8 WMMAs + branchless streaming update.
  // CAPTURE (compile-time) additionally records the diagonal sim value.
  auto process_tile = [&](const int ct, auto capC) {
    constexpr bool CAPTURE = decltype(capC)::value;
    const int col = (ct << 4) + l16;
    const _Float16* cp = eh + (size_t)col * DDIM;
    // Speculative prefetch of the next column tile (dropped if OOB).
    __builtin_prefetch(cp + 16 * DDIM, 0, 3);

    // B operand, same contiguous k order as A: lanes 0-15 hold N=lane, K=0..15; lanes 16-31
    // hold K=16..31. For a Gram matrix B's column data = row `col` of eh.
    v16h B[4];
#pragma unroll
    for (int ks = 0; ks < 4; ++ks)
      B[ks] = *(const v16h*)(cp + ks * 32 + half * 16);

    const float pj = props[col];

#pragma unroll
    for (int rt = 0; rt < 2; ++rt) {
      v8f acc = {};
#pragma unroll
      for (int ks = 0; ks < 4; ++ks)
        acc = __builtin_amdgcn_wmma_f32_16x16x32_f16(
            /*neg_a=*/false, A[rt][ks], /*neg_b=*/false, B[ks],
            /*c_mod=*/(short)0, acc, /*reuse_a=*/false, /*reuse_b=*/false);

      if constexpr (CAPTURE) {
        if (ct == dct0 + rt) {
#pragma unroll
          for (int r = 0; r < 8; ++r)
            if (laneDiag[r]) dval[rt] = acc[r];
        }
      }

      // acc[r] = cosine(row = rowBase+rt*16+half*8+r, col); logit = 10*acc[r]
#pragma unroll
      for (int r = 0; r < 8; ++r) {
        const float a = acc[r];
        z[rt][r] += EXP2F(fmaf(a, EXPK, -EXPK));            // exp(logit - 10)
        const float sel = (fabsf(prw[rt][r] - pj) < PTHRESH) ? 1.0f : 0.0f;
        c[rt][r] += sel;
        s[rt][r] = fmaf(sel, a, s[rt][r]);
      }
    }
  };

  using NoCap = std::integral_constant<bool, false>;
  using Cap   = std::integral_constant<bool, true>;

  const int ntiles = n >> 4;
  // Hot sweeps: zero diagonal logic, zero extra scalar branches.
  for (int ct = 0; ct < dct0; ++ct)          process_tile(ct, NoCap{});
  process_tile(dct0,     Cap{});             // rt=0 diagonal tile
  process_tile(dct0 + 1, Cap{});             // rt=1 diagonal tile
  for (int ct = dct0 + 2; ct < ntiles; ++ct) process_tile(ct, NoCap{});

  // Remove the diagonal's contribution (dp=0 so it passed the mask; the
  // reference sets logit_ii = -1e9 and mask_ii = 0).
#pragma unroll
  for (int rt = 0; rt < 2; ++rt) {
#pragma unroll
    for (int r = 0; r < 8; ++r) {
      if (laneDiag[r]) {
        const float a = dval[rt];
        z[rt][r] -= EXP2F(fmaf(a, EXPK, -EXPK));
        s[rt][r] -= a;
        c[rt][r] -= 1.0f;
      }
    }
  }

  // Reduce each accumulator across the 16 lanes of its half (columns).
#pragma unroll
  for (int rt = 0; rt < 2; ++rt) {
#pragma unroll
    for (int r = 0; r < 8; ++r) {
      float zz = z[rt][r], ss = s[rt][r], cc = c[rt][r];
#pragma unroll
      for (int msk = 1; msk < 16; msk <<= 1) {
        zz += __shfl_xor(zz, msk, 32);
        ss += __shfl_xor(ss, msk, 32);
        cc += __shfl_xor(cc, msk, 32);
      }
      if (l16 == r) {
        const int grow = rowBase + rt * 16 + half * 8 + r;
        const float lse = fmaf(LOG2F_RAW(zz), LN2, TEMP_INV); // 10 + ln(zz)
        const bool valid = cc > 0.0f;
        // per-row mean of -log_probs over positives: (c*lse - 10*s) / c
        per_row[grow] = valid ? (cc * lse - TEMP_INV * ss) / cc : 0.0f;
        validf[grow]  = valid ? 1.0f : 0.0f;
      }
    }
  }
}

// ---------------------------------------------------------------------------
// Kernel 3: deterministic final mean over valid rows (single block, fixed
// summation order -> identical result every replay).
// ---------------------------------------------------------------------------
__global__ void __launch_bounds__(256)
scl_reduce_kernel(const float* __restrict__ per_row,
                  const float* __restrict__ validf,
                  float* __restrict__ out, int n) {
  __shared__ float ss[256];
  __shared__ float sc[256];
  float s = 0.f, c = 0.f;
  for (int i = threadIdx.x; i < n; i += 256) { s += per_row[i]; c += validf[i]; }
  ss[threadIdx.x] = s; sc[threadIdx.x] = c;
  __syncthreads();
  for (int st = 128; st > 0; st >>= 1) {
    if ((int)threadIdx.x < st) {
      ss[threadIdx.x] += ss[threadIdx.x + st];
      sc[threadIdx.x] += sc[threadIdx.x + st];
    }
    __syncthreads();
  }
  if (threadIdx.x == 0) out[0] = (sc[0] > 0.f) ? (ss[0] / sc[0]) : 0.f;
}

// ---------------------------------------------------------------------------
extern "C" void kernel_launch(void* const* d_in, const int* in_sizes, int n_in,
                              void* d_out, int out_size, void* d_ws, size_t ws_size,
                              hipStream_t stream) {
  const float* emb   = (const float*)d_in[0];   // [N, 128] f32
  const float* props = (const float*)d_in[1];   // [N, 1]   f32
  const int n = in_sizes[0] / DDIM;             // 8192

  // Workspace: f16 normalized embeddings (2 MB) + per_row + valid flags.
  _Float16* eh   = (_Float16*)d_ws;
  float* per_row = (float*)((char*)d_ws + (size_t)n * DDIM * sizeof(_Float16));
  float* validf  = per_row + n;
  float* out     = (float*)d_out;

  scl_norm_f16_kernel<<<n, DDIM, 0, stream>>>(emb, eh);
  scl_sim_kernel<<<n / 32, 32, 0, stream>>>(eh, props, per_row, validf, n);
  scl_reduce_kernel<<<1, 256, 0, stream>>>(per_row, validf, out, n);
}
